// Hyper_Attention_54125177864651
// MI455X (gfx1250) — compile-verified, matrix operand order checked by reading, not run on hardware
//
#include <hip/hip_runtime.h>
#include <stdint.h>

// ---------------------------------------------------------------------------
// Hyper (Mahalanobis) attention for MI455X / gfx1250, wave32 + WMMA bf16.
// Heavy GEMMs: bf16 inputs, f32 accumulation via v_wmma_f32_16x16x32_bf16,
// tiles staged with GLOBAL_LOAD_ASYNC_TO_LDS_B128 (ASYNCcnt) + double buffer.
// ---------------------------------------------------------------------------

typedef __attribute__((ext_vector_type(16))) __bf16 v16bf;
typedef __attribute__((ext_vector_type(8)))  float  v8f;

union BF16Frag { v16bf v; uint4 u[2]; };

__device__ __forceinline__ unsigned short f2bf(float f) {
  unsigned int u = __float_as_uint(f);
  unsigned int r = 0x7FFFu + ((u >> 16) & 1u);   // round-to-nearest-even
  return (unsigned short)((u + r) >> 16);
}
__device__ __forceinline__ float bf2f(unsigned short h) {
  return __uint_as_float(((unsigned int)h) << 16);
}

// Async 16B copy global -> LDS (CDNA5 TDM-lite path, tracked by ASYNCcnt).
// LDS operand = low 32 bits of the generic address (ISA 10.2 aperture rule).
__device__ __forceinline__ void async_b128(unsigned lds_byte_addr,
                                           const unsigned short* g) {
  asm volatile("global_load_async_to_lds_b128 %0, %1, off"
               :: "v"(lds_byte_addr), "v"(g)
               : "memory");
}
__device__ __forceinline__ void wait_async0() {
  asm volatile("s_wait_asynccnt 0x0" ::: "memory");
}

// ---------------------------------------------------------------------------
// Batched GEMM: C[b][m][n] = sum_k A[b][m][k] * B[b][n][k]  (+ bias[n])
// A,B bf16 row-major (K contiguous), C f32.  Requires:
//   Mdim % 128 == 0, Ndim % BN == 0, Kdim % 32 == 0, lda/ldb % 8 == 0.
// Block: 256 thr = 8 wave32, wave grid 4(M) x 2(N); wave tile 32 x (BN/2).
// LDS double-buffered, filled by async loads; padded row stride 40 bf16
// (80 B = multiple of 16 B -> aligned b128 LDS ops, no bank conflicts).
// ---------------------------------------------------------------------------
template<int BN>
__global__ __launch_bounds__(256)
void gemm_bf16_wmma(const unsigned short* __restrict__ A, long long strideA, int lda,
                    const unsigned short* __restrict__ B, long long strideB, int ldb,
                    float* __restrict__ C, long long strideC, int ldc,
                    const float* __restrict__ bias,
                    int Mdim, int Ndim, int Kdim)
{
  constexpr int NJ    = BN / 32;        // N fragments per wave (2 or 4)
  constexpr int ELA   = 128 * 40;       // elements per A buffer
  constexpr int ELB   = BN * 40;        // elements per B buffer
  __shared__ __align__(16) unsigned short lsA[2 * ELA];
  __shared__ __align__(16) unsigned short lsB[2 * ELB];

  const int bz = blockIdx.z;
  A += (long long)bz * strideA;
  B += (long long)bz * strideB;
  C += (long long)bz * strideC;

  const int m0 = blockIdx.y * 128;
  const int n0 = blockIdx.x * BN;

  const int tid  = threadIdx.x;
  const int lane = tid & 31;
  const int wave = tid >> 5;
  const int wm   = wave & 3;            // wave tile along M (4 x 32)
  const int wn   = wave >> 2;           // wave tile along N (2 x BN/2)

  const unsigned lsA_base = (unsigned)(uintptr_t)(&lsA[0]);
  const unsigned lsB_base = (unsigned)(uintptr_t)(&lsB[0]);

  // per-thread tile-load coordinates (16B granules)
  const int arow = tid >> 2;            // A rows for s=0/1: arow, arow+64
  const int acol = (tid & 3) * 8;

  auto prefetch = [&](int kb, int buf) {
#pragma unroll
    for (int s = 0; s < 2; s++) {       // A: 128x32 = 512 granules
      int row = arow + s * 64;
      async_b128(lsA_base + (unsigned)(buf * ELA + row * 40 + acol) * 2,
                 A + (size_t)(m0 + row) * lda + kb + acol);
    }
#pragma unroll
    for (int s = 0; s < BN / 64; s++) { // B: BNx32
      int row = arow + s * 64;
      async_b128(lsB_base + (unsigned)(buf * ELB + row * 40 + acol) * 2,
                 B + (size_t)(n0 + row) * ldb + kb + acol);
    }
  };

  const v8f vzero = {0.f, 0.f, 0.f, 0.f, 0.f, 0.f, 0.f, 0.f};
  v8f acc[2][NJ];
#pragma unroll
  for (int i = 0; i < 2; i++)
#pragma unroll
    for (int j = 0; j < NJ; j++) acc[i][j] = vzero;

  const int hl  = lane >> 4;            // half-wave select (ISA 7.12.2 packing)
  const int l15 = lane & 15;

  prefetch(0, 0);
  wait_async0();
  __syncthreads();

  for (int kb = 0; kb < Kdim; kb += 32) {
    const int buf = (kb >> 5) & 1;
    if (kb + 32 < Kdim) prefetch(kb + 32, buf ^ 1);

    // fragments: lane L<16 holds K {0..7,16..23}, L>=16 holds K {8..15,24..31}
    BF16Frag af[2], bfr[NJ];
#pragma unroll
    for (int i = 0; i < 2; i++) {
      int r = buf * ELA + (wm * 32 + i * 16 + l15) * 40 + hl * 8;
      af[i].u[0] = *reinterpret_cast<const uint4*>(&lsA[r]);
      af[i].u[1] = *reinterpret_cast<const uint4*>(&lsA[r + 16]);
    }
#pragma unroll
    for (int j = 0; j < NJ; j++) {
      int r = buf * ELB + (wn * (BN / 2) + j * 16 + l15) * 40 + hl * 8;
      bfr[j].u[0] = *reinterpret_cast<const uint4*>(&lsB[r]);
      bfr[j].u[1] = *reinterpret_cast<const uint4*>(&lsB[r + 16]);
    }

#pragma unroll
    for (int i = 0; i < 2; i++)
#pragma unroll
      for (int j = 0; j < NJ; j++)
        acc[i][j] = __builtin_amdgcn_wmma_f32_16x16x32_bf16(
            false, af[i].v, false, bfr[j].v, (short)0, acc[i][j], false, false);

    wait_async0();
    __syncthreads();
  }

  // epilogue: 16x16 f32 C layout -> lane (hl,l15) holds rows hl*8..hl*8+7
  const int rq = hl * 8;
#pragma unroll
  for (int i = 0; i < 2; i++) {
#pragma unroll
    for (int j = 0; j < NJ; j++) {
      int cg = n0 + wn * (BN / 2) + j * 16 + l15;
      float bv = bias ? bias[cg] : 0.0f;
#pragma unroll
      for (int e = 0; e < 8; e++) {
        int rg = m0 + wm * 32 + i * 16 + rq + e;
        C[(size_t)rg * ldc + cg] = acc[i][j][e] + bv;
      }
    }
  }
}

// ---------------------------------------------------------------------------
// Small helper kernels
// ---------------------------------------------------------------------------
__global__ void k_f32_to_bf16(const float* __restrict__ in,
                              unsigned short* __restrict__ out, long long n) {
  long long i = (long long)blockIdx.x * blockDim.x + threadIdx.x;
  if (i < n) out[i] = f2bf(in[i]);
}

__global__ void k_transpose_f32_to_bf16(const float* __restrict__ in,
                                        unsigned short* __restrict__ out, int dim) {
  int i = blockIdx.x * blockDim.x + threadIdx.x;
  if (i < dim * dim) {
    int r = i / dim, c = i % dim;
    out[c * dim + r] = f2bf(in[r * dim + c]);
  }
}

__global__ void k_copy_f32(const float* __restrict__ in, float* __restrict__ out, int n) {
  int i = blockIdx.x * blockDim.x + threadIdx.x;
  if (i < n) out[i] = in[i];
}

// qkv f32 [B*N, 3C] -> q,k bf16 [B,H,N,D]; v transposed bf16 [B,H,D,N]
__global__ void k_split_qkv(const float* __restrict__ qkv,
                            unsigned short* __restrict__ q,
                            unsigned short* __restrict__ k,
                            unsigned short* __restrict__ vT) {
  long long idx = (long long)blockIdx.x * blockDim.x + threadIdx.x;
  const long long total = (long long)2 * 8 * 2048 * 192;
  if (idx >= total) return;
  int d = (int)(idx % 192);
  long long t = idx / 192;
  int n = (int)(t % 2048); t /= 2048;
  int h = (int)(t % 8);
  int b = (int)(t / 8);
  const float* row = qkv + ((long long)(b * 2048 + n)) * 4608 + h * 192 + d;
  q[idx] = f2bf(row[0]);
  k[idx] = f2bf(row[1536]);
  vT[(((long long)(b * 8 + h)) * 192 + d) * 2048 + n] = f2bf(row[3072]);
}

// out[row] = sum_d bf16(x[row,d]) * y[row,d]   (rows = B*H*N, D = 192)
__global__ __launch_bounds__(256)
void k_row_dot(const unsigned short* __restrict__ xb, const float* __restrict__ y,
               float* __restrict__ out, int rows) {
  int wave = threadIdx.x >> 5, lane = threadIdx.x & 31;
  long long row = (long long)blockIdx.x * 8 + wave;
  if (row >= rows) return;
  const unsigned short* xr = xb + row * 192;
  const float* yr = y + row * 192;
  float s = 0.f;
  for (int d = lane; d < 192; d += 32) s += bf2f(xr[d]) * yr[d];
#pragma unroll
  for (int off = 16; off > 0; off >>= 1) s += __shfl_xor(s, off, 32);
  if (lane == 0) out[row] = s;
}

// In-place row softmax of  scale*(2*qMk - qMq[n] - kMk[m]); also emit bf16 copy.
__global__ __launch_bounds__(256)
void k_softmax(float* __restrict__ attn, unsigned short* __restrict__ attn_bf,
               const float* __restrict__ qMq, const float* __restrict__ kMk,
               const float* __restrict__ scale_p) {
  const int n  = blockIdx.x;
  const int bh = blockIdx.y;
  const int tid = threadIdx.x;
  float* srow = attn + ((long long)bh * 2048 + n) * 2048;
  unsigned short* brow = attn_bf + ((long long)bh * 2048 + n) * 2048;
  const float* kk = kMk + (long long)bh * 2048;
  const float qq = qMq[(long long)bh * 2048 + n];
  const float sc = scale_p[0];

  float v[8];
  float mx = -3.4e38f;
#pragma unroll
  for (int i = 0; i < 8; i++) {
    int m = tid + i * 256;
    float logit = sc * (2.f * srow[m] - qq - kk[m]);
    v[i] = logit;
    mx = fmaxf(mx, logit);
  }
  __shared__ float red[8];
  int lane = tid & 31, wave = tid >> 5;
#pragma unroll
  for (int off = 16; off > 0; off >>= 1) mx = fmaxf(mx, __shfl_xor(mx, off, 32));
  if (lane == 0) red[wave] = mx;
  __syncthreads();
  mx = red[0];
#pragma unroll
  for (int w = 1; w < 8; w++) mx = fmaxf(mx, red[w]);
  __syncthreads();

  float sum = 0.f;
#pragma unroll
  for (int i = 0; i < 8; i++) { v[i] = __expf(v[i] - mx); sum += v[i]; }
#pragma unroll
  for (int off = 16; off > 0; off >>= 1) sum += __shfl_xor(sum, off, 32);
  if (lane == 0) red[wave] = sum;
  __syncthreads();
  float tot = 0.f;
#pragma unroll
  for (int w = 0; w < 8; w++) tot += red[w];
  float inv = 1.0f / tot;
#pragma unroll
  for (int i = 0; i < 8; i++) {
    int m = tid + i * 256;
    float a = v[i] * inv;
    srow[m] = a;
    brow[m] = f2bf(a);
  }
}

// out_h f32 [B,H,N,D] -> xo bf16 [B*N, C]
__global__ void k_merge_heads(const float* __restrict__ oh,
                              unsigned short* __restrict__ xo) {
  long long idx = (long long)blockIdx.x * blockDim.x + threadIdx.x;
  const long long total = (long long)2 * 8 * 2048 * 192;
  if (idx >= total) return;
  int d = (int)(idx % 192);
  long long t = idx / 192;
  int n = (int)(t % 2048); t /= 2048;
  int h = (int)(t % 8);
  int b = (int)(t / 8);
  xo[((long long)(b * 2048 + n)) * 1536 + h * 192 + d] = f2bf(oh[idx]);
}

// ---------------------------------------------------------------------------
// Host-side orchestration
// ---------------------------------------------------------------------------
extern "C" void kernel_launch(void* const* d_in, const int* in_sizes, int n_in,
                              void* d_out, int out_size, void* d_ws, size_t ws_size,
                              hipStream_t stream) {
  (void)in_sizes; (void)n_in; (void)out_size; (void)ws_size;

  const float* x      = (const float*)d_in[0];   // [2,2048,1536]
  const float* W_qkv  = (const float*)d_in[1];   // [4608,1536]
  const float* W_proj = (const float*)d_in[2];   // [1536,1536]
  const float* b_proj = (const float*)d_in[3];   // [1536]
  const float* scale  = (const float*)d_in[4];   // [1]
  const float* Mmat   = (const float*)d_in[5];   // [192,192]

  float* out_f  = (float*)d_out;                         // [4096,1536]
  float* attn_f = out_f + (long long)4096 * 1536;        // [16,2048,2048]
  float* M_out  = attn_f + (long long)16 * 2048 * 2048;  // [192,192]

  char* ws = (char*)d_ws;
  size_t off = 0;
  auto alloc = [&](size_t bytes) -> char* {
    char* p = ws + off;
    off += (bytes + 255) & ~(size_t)255;
    return p;
  };

  // persistent regions
  unsigned short* xb      = (unsigned short*)alloc((size_t)4096 * 1536 * 2);
  unsigned short* Wqkvb   = (unsigned short*)alloc((size_t)4608 * 1536 * 2);
  unsigned short* Wprojb  = (unsigned short*)alloc((size_t)1536 * 1536 * 2);
  unsigned short* Mtb     = (unsigned short*)alloc((size_t)192 * 192 * 2);
  unsigned short* q_bf    = (unsigned short*)alloc((size_t)16 * 2048 * 192 * 2);
  unsigned short* k_bf    = (unsigned short*)alloc((size_t)16 * 2048 * 192 * 2);
  unsigned short* vT_bf   = (unsigned short*)alloc((size_t)16 * 192 * 2048 * 2);
  unsigned short* qM_bf   = (unsigned short*)alloc((size_t)16 * 2048 * 192 * 2);
  float*          qMq     = (float*)alloc((size_t)16 * 2048 * 4);
  float*          kMk     = (float*)alloc((size_t)16 * 2048 * 4);
  unsigned short* attn_bf = (unsigned short*)alloc((size_t)16 * 2048 * 2048 * 2);
  // transient pool, reused phase by phase (max user: qkv_f32 = 75.5 MB)
  char* pool = alloc((size_t)4096 * 4608 * 4);
  float*          qkv_f = (float*)pool;                                   // phase A
  float*          qM_f  = (float*)pool;                                   // phase B
  float*          kM_f  = (float*)(pool + (size_t)16 * 2048 * 192 * 4);   // phase B
  float*          out_h = (float*)pool;                                   // phase C
  unsigned short* xo_bf = (unsigned short*)(pool + (size_t)16 * 2048 * 192 * 4); // phase C

  const int T = 256;
  const long long nQKVel = 6291456LL;   // B*H*N*D

  // 1) precision converts + M passthrough
  k_f32_to_bf16<<<(int)((6291456 + T - 1) / T), T, 0, stream>>>(x, xb, 6291456LL);
  k_f32_to_bf16<<<(int)((7077888 + T - 1) / T), T, 0, stream>>>(W_qkv, Wqkvb, 7077888LL);
  k_f32_to_bf16<<<(int)((2359296 + T - 1) / T), T, 0, stream>>>(W_proj, Wprojb, 2359296LL);
  k_transpose_f32_to_bf16<<<(36864 + T - 1) / T, T, 0, stream>>>(Mmat, Mtb, 192);
  k_copy_f32<<<(36864 + T - 1) / T, T, 0, stream>>>(Mmat, M_out, 36864);

  // 2) qkv = x @ W_qkv^T : [4096,1536] x [4608,1536]^T
  gemm_bf16_wmma<128><<<dim3(36, 32, 1), T, 0, stream>>>(
      xb, 0, 1536, Wqkvb, 0, 1536, qkv_f, 0, 4608, nullptr, 4096, 4608, 1536);
  k_split_qkv<<<(int)((nQKVel + T - 1) / T), T, 0, stream>>>(qkv_f, q_bf, k_bf, vT_bf);

  // 3) qM = q @ M, kM = k @ M   (batched over B*H, B-matrix = M^T shared)
  gemm_bf16_wmma<64><<<dim3(3, 16, 16), T, 0, stream>>>(
      q_bf, 393216, 192, Mtb, 0, 192, qM_f, 393216, 192, nullptr, 2048, 192, 192);
  gemm_bf16_wmma<64><<<dim3(3, 16, 16), T, 0, stream>>>(
      k_bf, 393216, 192, Mtb, 0, 192, kM_f, 393216, 192, nullptr, 2048, 192, 192);

  // 4) qMq, kMk row dots; bf16 copy of qM for the score GEMM
  k_row_dot<<<4096, T, 0, stream>>>(q_bf, qM_f, qMq, 32768);
  k_row_dot<<<4096, T, 0, stream>>>(k_bf, kM_f, kMk, 32768);
  k_f32_to_bf16<<<(int)((nQKVel + T - 1) / T), T, 0, stream>>>(qM_f, qM_bf, nQKVel);

  // 5) scores: qMk = qM @ k^T  -> straight into d_out attn region (f32)
  gemm_bf16_wmma<128><<<dim3(16, 16, 16), T, 0, stream>>>(
      qM_bf, 393216, 192, k_bf, 393216, 192,
      attn_f, 4194304, 2048, nullptr, 2048, 2048, 192);

  // 6) softmax(-dists*scale) in place; also bf16 attn for next GEMM
  k_softmax<<<dim3(2048, 16, 1), T, 0, stream>>>(attn_f, attn_bf, qMq, kMk, scale);

  // 7) out_h = attn @ v   (B-matrix = v^T, K contiguous)
  gemm_bf16_wmma<64><<<dim3(3, 16, 16), T, 0, stream>>>(
      attn_bf, 4194304, 2048, vT_bf, 393216, 2048,
      out_h, 393216, 192, nullptr, 2048, 192, 2048);
  k_merge_heads<<<(int)((nQKVel + T - 1) / T), T, 0, stream>>>(out_h, xo_bf);

  // 8) out = xo @ W_proj^T + b_proj  -> d_out head
  gemm_bf16_wmma<128><<<dim3(12, 32, 1), T, 0, stream>>>(
      xo_bf, 0, 1536, Wprojb, 0, 1536, out_f, 0, 1536, b_proj, 4096, 1536, 1536);
}
